// EncoderAttention3_24919400252111
// MI455X (gfx1250) — compile-verified, matrix operand order checked by reading, not run on hardware
//
#include <hip/hip_runtime.h>
#include <hip/hip_bf16.h>

// ---------------------------------------------------------------------------
// EncoderAttention3 on MI455X (gfx1250): f16 WMMA pipeline, f32 accumulate.
// out[b] = concat(obs_ego[b] (48), sum_n MLP(x[b,n]) (64)),  B=16384
// MLP: 73 -> 256 (LN+tanh) -> 256 (ELU) -> 64
// ---------------------------------------------------------------------------

typedef _Float16 v8h  __attribute__((ext_vector_type(8)));
typedef _Float16 v16h __attribute__((ext_vector_type(16)));
typedef float    v8f  __attribute__((ext_vector_type(8)));

#define BATCH      16384
#define OBS_DIM    216
#define EGO        48
#define NEI        7
#define K1P        96    // 73 padded to 3 k-steps of 32
#define HID        256
#define DOUT       64
#define OUT_DIM    112   // 48 + 64

// LDS strides (in halves / floats) padded for bank-conflict avoidance
#define SX   104   // X row stride, halves   (96 + 8)
#define S1F  260   // h1 f32 row stride, floats
#define S1H  264   // h1 f16 row stride, halves
#define S2H  264   // h2 f16 row stride, halves

#define W1T_ELEMS (HID * K1P)     // 24576
#define W2T_ELEMS (HID * HID)     // 65536
#define W3T_ELEMS (DOUT * HID)    // 16384
#define WS_BYTES  ((W1T_ELEMS + W2T_ELEMS + W3T_ELEMS) * 2)  // 212992

// Load 16 f16 in the ISA-specified WMMA A/B per-lane layout:
// lane covers K chunks [k0+8*half, +8) and [k0+16+8*half, +8)
static __device__ __forceinline__ v16h load_frag(const _Float16* p) {
    v8h lo = *(const v8h*)p;
    v8h hi = *(const v8h*)(p + 16);
    return __builtin_shufflevector(lo, hi, 0,1,2,3,4,5,6,7,8,9,10,11,12,13,14,15);
}

static __device__ __forceinline__ float fast_tanh(float y) {
#if __has_builtin(__builtin_amdgcn_tanhf)
    return __builtin_amdgcn_tanhf(y);       // gfx1250 native v_tanh_f32
#else
    // tanh(y) = 1 - 2/(exp(2y)+1); saturates correctly at +-1
    return 1.0f - 2.0f / (__expf(2.0f * y) + 1.0f);
#endif
}

// ---------------------------------------------------------------------------
// Prep: transpose weights to f16 row-major [N][K] strips (B-matrix friendly).
// ---------------------------------------------------------------------------
__global__ void prep_weights(const float* __restrict__ W1,
                             const float* __restrict__ W2,
                             const float* __restrict__ W3,
                             _Float16* __restrict__ w1t,
                             _Float16* __restrict__ w2t,
                             _Float16* __restrict__ w3t) {
    int i = blockIdx.x * 256 + threadIdx.x;
    if (i < W1T_ELEMS) {
        int o = i / K1P, k = i % K1P;                 // W1: (73,256) -> [256][96]
        w1t[i] = (k < 73) ? (_Float16)W1[k * HID + o] : (_Float16)0.0f;
    } else if (i < W1T_ELEMS + W2T_ELEMS) {
        int j = i - W1T_ELEMS;
        int o = j >> 8, k = j & 255;                  // W2: (256,256) -> [256][256]
        w2t[j] = (_Float16)W2[k * HID + o];
    } else if (i < W1T_ELEMS + W2T_ELEMS + W3T_ELEMS) {
        int j = i - W1T_ELEMS - W2T_ELEMS;
        int o = j >> 8, k = j & 255;                  // W3: (256,64) -> [64][256]
        w3t[j] = (_Float16)W3[k * DOUT + o];
    }
}

// ---------------------------------------------------------------------------
// Main kernel: one block = 16 batches, 4 waves, 7 neighbor row-tiles of 16.
// ---------------------------------------------------------------------------
__global__ __launch_bounds__(128, 1)
void encoder_attention3_kernel(const float* __restrict__ obs,
                               const float* __restrict__ b1,
                               const float* __restrict__ lng,
                               const float* __restrict__ lnb,
                               const float* __restrict__ b2,
                               const float* __restrict__ b3,
                               const _Float16* __restrict__ w1t,
                               const _Float16* __restrict__ w2t,
                               const _Float16* __restrict__ w3t,
                               float* __restrict__ out) {
    __shared__ __align__(16) _Float16 Xs[NEI * 16 * SX];   // 23296 B
    __shared__ __align__(16) float    H1[16 * S1F];        // 16640 B
    __shared__ __align__(16) _Float16 H1h[16 * S1H];       //  8448 B
    __shared__ __align__(16) _Float16 H2h[16 * S2H];       //  8448 B
    __shared__              float    red[2][16][8];        //  1024 B

    const int tid  = threadIdx.x;
    const int lane = tid & 31;
    const int wave = tid >> 5;
    const int m    = lane & 15;   // row (A) or N-col (B/D) within tile
    const int half = lane >> 4;
    const int b0   = blockIdx.x * 16;

    // ---- build X in LDS: row r = n*16 + bl, f16, zero-padded K 73..95 ----
    if (tid < NEI * 16) {
        const int n  = tid >> 4;
        const int bl = tid & 15;
        const float* o = obs + (b0 + bl) * OBS_DIM;
        _Float16* xr = Xs + tid * SX;
        for (int i = 0; i < EGO; ++i) xr[i] = (_Float16)o[i];
        for (int j = 0; j < 24; ++j) xr[EGO + j] = (_Float16)o[EGO + j * NEI + n];
        xr[72] = (n >= 3) ? (_Float16)1.0f : (_Float16)0.0f;   // TEAM_IDS[1:][n]
        for (int k = 73; k < K1P; ++k) xr[k] = (_Float16)0.0f;
    }

    // ---- ego passthrough ----
    for (int i = tid; i < 16 * EGO; i += 128) {
        int bl = i / EGO, c = i % EGO;
        out[(b0 + bl) * OUT_DIM + c] = obs[(b0 + bl) * OBS_DIM + c];
    }

    // ---- warm L2 for the weight strips (gfx1250 global_prefetch_b8) ----
    {
        const char* wp = (const char*)w1t;
        for (int off = tid * 256; off < WS_BYTES; off += 128 * 256)
            __builtin_prefetch(wp + off, 0, 3);
    }

    v8f acc3 = {};   // L3 accumulator: carries k-loop AND neighbor-sum

    __syncthreads();

    for (int t = 0; t < NEI; ++t) {
        const _Float16* Xt = Xs + t * 16 * SX;

        // -------- L1: X(16x96) @ W1 -> H1 f32 (waves split 16 col tiles) ---
        for (int cc = 0; cc < 4; ++cc) {
            const int c = wave + cc * 4;
            v8f C = {};
            #pragma unroll
            for (int kk = 0; kk < 3; ++kk) {
                const int k0 = kk * 32;
                v16h A  = load_frag(Xt + m * SX + k0 + half * 8);
                v16h Bm = load_frag(w1t + (c * 16 + m) * K1P + k0 + half * 8);
                C = __builtin_amdgcn_wmma_f32_16x16x32_f16(
                        false, A, false, Bm, (short)0, C, false, false);
            }
            const int col = c * 16 + m;
            #pragma unroll
            for (int vi = 0; vi < 8; ++vi)
                H1[(vi + 8 * half) * S1F + col] = C[vi];
        }
        __syncthreads();

        // -------- +b1, LayerNorm, tanh -> H1h f16 --------------------------
        {
            const int r = tid >> 3, j = tid & 7, cb = j * 32;
            const float* hr = H1 + r * S1F + cb;
            float v[32], s = 0.0f, s2 = 0.0f;
            #pragma unroll
            for (int i = 0; i < 32; ++i) {
                float x = hr[i] + b1[cb + i];
                v[i] = x; s += x; s2 += x * x;
            }
            red[0][r][j] = s; red[1][r][j] = s2;
            __syncthreads();
            float ts = 0.0f, ts2 = 0.0f;
            #pragma unroll
            for (int j2 = 0; j2 < 8; ++j2) { ts += red[0][r][j2]; ts2 += red[1][r][j2]; }
            const float mu  = ts * (1.0f / 256.0f);
            const float var = ts2 * (1.0f / 256.0f) - mu * mu;
            const float inv = __frsqrt_rn(var + 1e-5f);
            _Float16* dst = H1h + r * S1H + cb;
            #pragma unroll
            for (int i = 0; i < 32; ++i) {
                float y = (v[i] - mu) * inv * lng[cb + i] + lnb[cb + i];
                dst[i] = (_Float16)fast_tanh(y);
            }
        }
        __syncthreads();

        // -------- L2: h1(16x256) @ W2, +b2, ELU -> H2h f16 -----------------
        for (int cc = 0; cc < 4; ++cc) {
            const int c = wave + cc * 4;
            v8f C = {};
            #pragma unroll
            for (int kk = 0; kk < 8; ++kk) {
                const int k0 = kk * 32;
                v16h A  = load_frag(H1h + m * S1H + k0 + half * 8);
                v16h Bm = load_frag(w2t + (c * 16 + m) * HID + k0 + half * 8);
                C = __builtin_amdgcn_wmma_f32_16x16x32_f16(
                        false, A, false, Bm, (short)0, C, false, false);
            }
            const int col = c * 16 + m;
            const float bv = b2[col];
            #pragma unroll
            for (int vi = 0; vi < 8; ++vi) {
                float x = C[vi] + bv;
                float e = (x > 0.0f) ? x : (__expf(x) - 1.0f);   // ELU
                H2h[(vi + 8 * half) * S2H + col] = (_Float16)e;
            }
        }
        __syncthreads();

        // -------- L3: h2(16x256) @ W3; accumulate over k AND neighbors -----
        {
            const int c = wave;   // 4 waves x 16 cols = 64 outputs
            #pragma unroll
            for (int kk = 0; kk < 8; ++kk) {
                const int k0 = kk * 32;
                v16h A  = load_frag(H2h + m * S2H + k0 + half * 8);
                v16h Bm = load_frag(w3t + (c * 16 + m) * HID + k0 + half * 8);
                acc3 = __builtin_amdgcn_wmma_f32_16x16x32_f16(
                           false, A, false, Bm, (short)0, acc3, false, false);
            }
        }
        __syncthreads();
    }

    // ---- +b3, store latent: D rows are batches (same batch across tiles) --
    {
        const int col = wave * 16 + m;
        const float bv = b3[col];
        #pragma unroll
        for (int vi = 0; vi < 8; ++vi) {
            const int bl = vi + 8 * half;
            out[(b0 + bl) * OUT_DIM + EGO + col] = acc3[vi] + bv;
        }
    }
}

// ---------------------------------------------------------------------------
extern "C" void kernel_launch(void* const* d_in, const int* in_sizes, int n_in,
                              void* d_out, int out_size, void* d_ws, size_t ws_size,
                              hipStream_t stream) {
    const float* obs = (const float*)d_in[0];
    const float* W1  = (const float*)d_in[1];
    const float* b1  = (const float*)d_in[2];
    const float* lng = (const float*)d_in[3];
    const float* lnb = (const float*)d_in[4];
    const float* W2  = (const float*)d_in[5];
    const float* b2  = (const float*)d_in[6];
    const float* W3  = (const float*)d_in[7];
    const float* b3  = (const float*)d_in[8];
    float* out = (float*)d_out;

    _Float16* w1t = (_Float16*)d_ws;
    _Float16* w2t = w1t + W1T_ELEMS;
    _Float16* w3t = w2t + W2T_ELEMS;

    const int prep_total = W1T_ELEMS + W2T_ELEMS + W3T_ELEMS;
    prep_weights<<<(prep_total + 255) / 256, 256, 0, stream>>>(W1, W2, W3, w1t, w2t, w3t);

    encoder_attention3_kernel<<<BATCH / 16, 128, 0, stream>>>(
        obs, b1, lng, lnb, b2, b3, w1t, w2t, w3t, out);
}
